// IndependentSemanticCodebooks_75849122447657
// MI455X (gfx1250) — matrix operand order repaired and checked by reading, not compiled, not run on hardware
//
#include <hip/hip_runtime.h>

typedef __attribute__((ext_vector_type(16))) _Float16 v16h;
typedef __attribute__((ext_vector_type(8)))  float    v8f;

#define NG 5
#define NB 131072
#define NK 512
#define ND 64

#define WAVES 8
#define ITERS 4
#define ROWS_PER_PASS 32                              // 2 B-tiles of 16 rows per wave pass
#define ROWS_PER_BLOCK (WAVES * ROWS_PER_PASS * ITERS)   // 1024
#define BLOCKS_PER_GROUP (NB / ROWS_PER_BLOCK)        // 128
#define NBLOCKS (NG * BLOCKS_PER_GROUP)               // 640

#define ROWH 72   // padded halves per codebook row in LDS (144 B: 16B-aligned, conflict-free b128)

// output layout (floats): quantized [G*B*D] | per_group_loss [5] | total_loss [1] | indices [G*B]
#define QB_ELEMS ((size_t)NG * NB * ND)               // 41943040
#define PG_OFF   (QB_ELEMS)
#define TOT_OFF  (QB_ELEMS + NG)
#define IDX_OFF  (QB_ELEMS + NG + 1)

#define LOSS_SCALE (1.25f / ((float)NB * (float)ND))

__global__ __launch_bounds__(256, 1)
void vq_main(const float* __restrict__ feats,
             const float* __restrict__ cbs,
             float* __restrict__ out,
             float* __restrict__ ws)   // per-block partials, or nullptr -> atomic fallback
{
    __shared__ _Float16 lds_cb[NK * ROWH];
    __shared__ float    lds_esq[NK];
    __shared__ float    lds_wssd[WAVES];

    const int tid  = threadIdx.x;
    const int g    = blockIdx.x / BLOCKS_PER_GROUP;
    const int blk  = blockIdx.x % BLOCKS_PER_GROUP;
    const float* cb = cbs + (size_t)g * NK * ND;

    // ---- stage codebook into LDS as f16 (padded rows) + (-0.5*||e||^2) per code ----
    for (int r = tid; r < NK; r += 256) {
        const float* row = cb + (size_t)r * ND;
        float ss = 0.f;
        #pragma unroll
        for (int d4 = 0; d4 < ND; d4 += 4) {
            float4 v = *(const float4*)(row + d4);
            ss += v.x * v.x + v.y * v.y + v.z * v.z + v.w * v.w;
            lds_cb[r * ROWH + d4 + 0] = (_Float16)v.x;
            lds_cb[r * ROWH + d4 + 1] = (_Float16)v.y;
            lds_cb[r * ROWH + d4 + 2] = (_Float16)v.z;
            lds_cb[r * ROWH + d4 + 3] = (_Float16)v.w;
        }
        lds_esq[r] = -0.5f * ss;
    }
    __syncthreads();

    const int wave  = tid >> 5;
    const int lane  = tid & 31;
    const int lhalf = lane >> 4;   // 0 or 1
    const int lmod  = lane & 15;

    float ssd_acc = 0.f;

    for (int it = 0; it < ITERS; ++it) {
        const int rowBase = blk * ROWS_PER_BLOCK + it * (WAVES * ROWS_PER_PASS) + wave * ROWS_PER_PASS;

        if (it + 1 < ITERS)
            __builtin_prefetch(feats + ((size_t)g * NB + rowBase + lmod) * ND
                                     + (size_t)WAVES * ROWS_PER_PASS * ND, 0, 0);

        // ---- build B fragments for two 16-row tiles (features, f32 -> f16) ----
        v16h bfrag[2][2];
        #pragma unroll
        for (int t = 0; t < 2; ++t) {
            const float* frow = feats + ((size_t)g * NB + rowBase + t * 16 + lmod) * ND;
            #pragma unroll
            for (int kt = 0; kt < 2; ++kt) {
                const int dbase = kt * 32 + lhalf * 16;
                union { float4 f4[4]; float f[16]; } u;
                u.f4[0] = *(const float4*)(frow + dbase + 0);
                u.f4[1] = *(const float4*)(frow + dbase + 4);
                u.f4[2] = *(const float4*)(frow + dbase + 8);
                u.f4[3] = *(const float4*)(frow + dbase + 12);
                #pragma unroll
                for (int j = 0; j < 16; ++j)
                    bfrag[t][kt][j] = (_Float16)u.f[j];
            }
        }

        // ---- scan all 512 codes: score = dot(code, feat) - 0.5*||code||^2 ; argmax ----
        float bestv[2] = { -3.402823466e38f, -3.402823466e38f };
        int   besti[2] = { 0, 0 };

        #pragma unroll 2
        for (int nt = 0; nt < NK / 16; ++nt) {
            const int rowc  = nt * 16 + lmod;       // code row this lane fetches (A fragment)
            const int hoff  = lhalf * 8;
            const int ebase = nt * 16 + hoff;       // code indices this lane accumulates

            v8f einit;
            #pragma unroll
            for (int r = 0; r < 8; ++r) einit[r] = lds_esq[ebase + r];

            union { uint4 u[2]; v16h h; } a0, a1;
            const _Float16* ap = &lds_cb[rowc * ROWH + lhalf * 16];
            a0.u[0] = *(const uint4*)(ap);
            a0.u[1] = *(const uint4*)(ap + 8);
            a1.u[0] = *(const uint4*)(ap + 32);
            a1.u[1] = *(const uint4*)(ap + 40);

            // two independent accumulator chains: hides WMMA->WMMA RAW hazard
            v8f acc0 = __builtin_amdgcn_wmma_f32_16x16x32_f16(
                         false, a0.h, false, bfrag[0][0], (short)0, einit, false, false);
            v8f acc1 = __builtin_amdgcn_wmma_f32_16x16x32_f16(
                         false, a0.h, false, bfrag[1][0], (short)0, einit, false, false);
            acc0 = __builtin_amdgcn_wmma_f32_16x16x32_f16(
                         false, a1.h, false, bfrag[0][1], (short)0, acc0, false, false);
            acc1 = __builtin_amdgcn_wmma_f32_16x16x32_f16(
                         false, a1.h, false, bfrag[1][1], (short)0, acc1, false, false);

            #pragma unroll
            for (int r = 0; r < 8; ++r) {
                const int idx = ebase + r;
                float v0 = acc0[r];
                bool  b0 = v0 > bestv[0];           // strict: keeps lowest index on ties
                bestv[0] = b0 ? v0 : bestv[0];
                besti[0] = b0 ? idx : besti[0];
                float v1 = acc1[r];
                bool  b1 = v1 > bestv[1];
                bestv[1] = b1 ? v1 : bestv[1];
                besti[1] = b1 ? idx : besti[1];
            }
        }

        // ---- per tile: combine lane-halves, emit index, gather code row, accumulate SSD ----
        #pragma unroll
        for (int t = 0; t < 2; ++t) {
            float bv = bestv[t];
            int   bi = besti[t];
            float ov = __shfl_xor(bv, 16, 32);
            int   oi = __shfl_xor(bi, 16, 32);
            bool take = (ov > bv) || ((ov == bv) && (oi < bi));
            bi = take ? oi : bi;

            const int rowIdx = rowBase + t * 16 + lmod;
            if (lane < 16)
                out[IDX_OFF + (size_t)g * NB + rowIdx] = (float)bi;

            const float* qrow = cb + (size_t)bi * ND;
            const float* frow = feats + ((size_t)g * NB + rowIdx) * ND;   // L2-hot reload
            float*       orow = out + ((size_t)g * NB + rowIdx) * ND;
            #pragma unroll
            for (int kt = 0; kt < 2; ++kt) {
                const int dbase = kt * 32 + lhalf * 16;
                #pragma unroll
                for (int j4 = 0; j4 < 16; j4 += 4) {
                    float4 q = *(const float4*)(qrow + dbase + j4);
                    float4 f = *(const float4*)(frow + dbase + j4);
                    *(float4*)(orow + dbase + j4) = q;
                    float dx = q.x - f.x;
                    float dy = q.y - f.y;
                    float dz = q.z - f.z;
                    float dw = q.w - f.w;
                    ssd_acc += dx * dx + dy * dy + dz * dz + dw * dw;
                }
            }
        }
    }

    // ---- reduce SSD: wave -> block ----
    #pragma unroll
    for (int m = 1; m < 32; m <<= 1) ssd_acc += __shfl_xor(ssd_acc, m, 32);
    if (lane == 0) lds_wssd[wave] = ssd_acc;
    __syncthreads();
    if (tid == 0) {
        float s = 0.f;
        #pragma unroll
        for (int w = 0; w < WAVES; ++w) s += lds_wssd[w];
        if (ws) {
            ws[blockIdx.x] = s;                       // deterministic path
        } else {
            float v = s * LOSS_SCALE;                 // fallback (pre-zeroed slots)
            atomicAdd(&out[PG_OFF + g], v);
            atomicAdd(&out[TOT_OFF], v);
        }
    }
}

__global__ void vq_reduce(const float* __restrict__ ws, float* __restrict__ out)
{
    __shared__ float pg[NG];
    const int t = threadIdx.x;
    if (t < NG) {
        float s = 0.f;
        for (int i = 0; i < BLOCKS_PER_GROUP; ++i)
            s += ws[t * BLOCKS_PER_GROUP + i];        // fixed order -> deterministic
        float v = s * LOSS_SCALE;
        out[PG_OFF + t] = v;
        pg[t] = v;
    }
    __syncthreads();
    if (t == 0) {
        float s = 0.f;
        for (int gg = 0; gg < NG; ++gg) s += pg[gg];
        out[TOT_OFF] = s;
    }
}

__global__ void vq_zero_loss(float* __restrict__ out)
{
    if (threadIdx.x < NG + 1) out[PG_OFF + threadIdx.x] = 0.f;
}

extern "C" void kernel_launch(void* const* d_in, const int* in_sizes, int n_in,
                              void* d_out, int out_size, void* d_ws, size_t ws_size,
                              hipStream_t stream)
{
    const float* feats = (const float*)d_in[0];   // [G, B, D] f32
    const float* cbs   = (const float*)d_in[1];   // [G, K, D] f32
    float* out = (float*)d_out;
    float* ws  = (float*)d_ws;

    const bool use_ws = (ws != nullptr) && (ws_size >= (size_t)NBLOCKS * sizeof(float));

    if (!use_ws)
        vq_zero_loss<<<1, 32, 0, stream>>>(out);

    vq_main<<<NBLOCKS, 256, 0, stream>>>(feats, cbs, out, use_ws ? ws : nullptr);

    if (use_ws)
        vq_reduce<<<1, 64, 0, stream>>>(ws, out);
}
